// TemporalSNNClassifier_52931176956432
// MI455X (gfx1250) — matrix operand order repaired and checked by reading, not compiled, not run on hardware
//
#include <hip/hip_runtime.h>

typedef _Float16 f16;
typedef __attribute__((ext_vector_type(16))) _Float16 v16h;
typedef __attribute__((ext_vector_type(8)))  _Float16 v8h;
typedef __attribute__((ext_vector_type(8)))  float    v8f;

#define BATCH   16384
#define IN_DIM  256
#define HID     128
#define NCLS    8
#define TSTEPS  64
#define BETA    0.9f
#define THRESH  1.0f

#define WAVES_PER_BLOCK 4
#define LDS_PITCH 144   // f16 elems per batch-row: 128 + 16 pad -> 16B-aligned, bank-conflict-free

// convert 16 consecutive floats -> v16h (K = 16 consecutive values per lane)
__device__ __forceinline__ v16h cvt16(const float* __restrict__ p) {
    float4 f0 = ((const float4*)p)[0];
    float4 f1 = ((const float4*)p)[1];
    float4 f2 = ((const float4*)p)[2];
    float4 f3 = ((const float4*)p)[3];
    v16h r;
    r[0]=(f16)f0.x; r[1]=(f16)f0.y; r[2]=(f16)f0.z; r[3]=(f16)f0.w;
    r[4]=(f16)f1.x; r[5]=(f16)f1.y; r[6]=(f16)f1.z; r[7]=(f16)f1.w;
    r[8]=(f16)f2.x; r[9]=(f16)f2.y; r[10]=(f16)f2.z; r[11]=(f16)f2.w;
    r[12]=(f16)f3.x; r[13]=(f16)f3.y; r[14]=(f16)f3.z; r[15]=(f16)f3.w;
    return r;
}

// convert two disjoint 8-float chunks -> v16h (A-frag: K = {base..base+7, base+16..base+23})
__device__ __forceinline__ v16h cvt8x2(const float* __restrict__ p0, const float* __restrict__ p1) {
    float4 f0 = ((const float4*)p0)[0];
    float4 f1 = ((const float4*)p0)[1];
    float4 f2 = ((const float4*)p1)[0];
    float4 f3 = ((const float4*)p1)[1];
    v16h r;
    r[0]=(f16)f0.x; r[1]=(f16)f0.y; r[2]=(f16)f0.z; r[3]=(f16)f0.w;
    r[4]=(f16)f1.x; r[5]=(f16)f1.y; r[6]=(f16)f1.z; r[7]=(f16)f1.w;
    r[8]=(f16)f2.x; r[9]=(f16)f2.y; r[10]=(f16)f2.z; r[11]=(f16)f2.w;
    r[12]=(f16)f3.x; r[13]=(f16)f3.y; r[14]=(f16)f3.z; r[15]=(f16)f3.w;
    return r;
}

__global__ __launch_bounds__(128, 1) void snn_fused_kernel(
    const float* __restrict__ x,  const float* __restrict__ W1,
    const float* __restrict__ b1, const float* __restrict__ W2,
    const float* __restrict__ b2, float* __restrict__ out)
{
    __shared__ f16 lds_spk[WAVES_PER_BLOCK][16 * LDS_PITCH];

    const int lane    = threadIdx.x & 31;
    const int wave    = threadIdx.x >> 5;
    const int col     = lane & 15;   // batch row within tile (N-col of GEMM1^T, A-row of GEMM2)
    const int half    = lane >> 4;   // lane half selects K/M sub-chunk per WMMA layouts
    const int rowBase = (blockIdx.x * WAVES_PER_BLOCK + wave) * 16;

    f16* myLds = &lds_spk[wave][0];

    // ---- GEMM1 (transposed): cur1T[128 x 16] = W1[128x256] @ xT_tile[256x16] + b1 ----
    // C-layout per h-tile ht: lane holds batch col m=col; VGPR i -> h = ht*16 + half*8 + i
    v8f c1t[8];
    #pragma unroll
    for (int ht = 0; ht < 8; ++ht) {
        float4 bb0 = *(const float4*)(b1 + ht * 16 + half * 8);
        float4 bb1 = *(const float4*)(b1 + ht * 16 + half * 8 + 4);
        c1t[ht][0] = bb0.x; c1t[ht][1] = bb0.y; c1t[ht][2] = bb0.z; c1t[ht][3] = bb0.w;
        c1t[ht][4] = bb1.x; c1t[ht][5] = bb1.y; c1t[ht][6] = bb1.z; c1t[ht][7] = bb1.w;
    }

    const float* xrow = x + (size_t)(rowBase + col) * IN_DIM;
    #pragma unroll 1
    for (int kt = 0; kt < 8; ++kt) {
        const int kbase = kt * 32;
        // B fragment (x^T, 32x16): lane col = batch row; K = kbase + half*8 + {0..7}, kbase + 16 + half*8 + {0..7}
        v16h bfrag = cvt8x2(xrow + kbase + half * 8, xrow + kbase + 16 + half * 8);
        #pragma unroll
        for (int ht = 0; ht < 8; ++ht) {
            // A fragment (W1, 16x32): lane row = h = ht*16 + col;
            // VGPR0-3: K=kbase+half*8+{0..7}, VGPR4-7: K=kbase+16+half*8+{0..7}
            const float* w1r = W1 + (size_t)(ht * 16 + col) * IN_DIM;
            v16h afrag = cvt8x2(w1r + kbase + half * 8,
                                w1r + kbase + 16 + half * 8);
            c1t[ht] = __builtin_amdgcn_wmma_f32_16x16x32_f16(
                false, afrag, false, bfrag, (short)0, c1t[ht], false, false);
        }
    }

    // ---- resident W2 B-fragments (K=128 -> 4 tiles), N padded 8->16 ----
    v16h w2f[4];
    #pragma unroll
    for (int kt = 0; kt < 4; ++kt) {
        if (col < NCLS) {
            w2f[kt] = cvt8x2(W2 + (size_t)col * HID + kt * 32 + half * 8, W2 + (size_t)col * HID + kt * 32 + 16 + half * 8);
        } else {
            v16h z;
            #pragma unroll
            for (int i = 0; i < 16; ++i) z[i] = (f16)0.0f;
            w2f[kt] = z;
        }
    }
    const float b2v = (col < NCLS) ? b2[col] : 0.0f;

    // ---- temporal LIF loop ----
    v8f mem1t[8];
    #pragma unroll
    for (int ht = 0; ht < 8; ++ht)
        #pragma unroll
        for (int i = 0; i < 8; ++i) mem1t[ht][i] = 0.0f;

    v8f mem2, acc;
    #pragma unroll
    for (int i = 0; i < 8; ++i) { mem2[i] = 0.0f; acc[i] = 0.0f; }

    #pragma unroll 1
    for (int t = 0; t < TSTEPS; ++t) {
        // LIF layer 1 in transposed layout; one packed b128 spike store per h-tile.
        // Spikes land row-major: spk[m][h], m = col, h = ht*16 + half*8 + i
        #pragma unroll
        for (int ht = 0; ht < 8; ++ht) {
            v8h s;
            #pragma unroll
            for (int i = 0; i < 8; ++i) {
                float mo  = mem1t[ht][i];
                float rst = (mo > THRESH) ? THRESH : 0.0f;
                float mn  = fmaf(BETA, mo, c1t[ht][i]) - rst;
                mem1t[ht][i] = mn;
                s[i] = (mn > THRESH) ? (f16)1.0f : (f16)0.0f;
            }
            *(v8h*)(myLds + col * LDS_PITCH + ht * 16 + half * 8) = s;
        }
        asm volatile("s_wait_dscnt 0" ::: "memory");  // spike stores visible before A-frag reads

        // GEMM2: cur2[16x16] = spk1[16x128] @ W2^T (cols 8..15 zero padding)
        v8f c2;
        #pragma unroll
        for (int i = 0; i < 8; ++i) c2[i] = b2v;
        #pragma unroll
        for (int kt = 0; kt < 4; ++kt) {
            const int kbase = kt * 32;
            // Row-major A-frag read: lane row m = col
            v8h a0 = *(const v8h*)(myLds + col * LDS_PITCH + kbase + half * 8);
            v8h a1 = *(const v8h*)(myLds + col * LDS_PITCH + kbase + 16 + half * 8);
            v16h afrag;
            #pragma unroll
            for (int i = 0; i < 8; ++i) { afrag[i] = a0[i]; afrag[8 + i] = a1[i]; }
            c2 = __builtin_amdgcn_wmma_f32_16x16x32_f16(
                false, afrag, false, w2f[kt], (short)0, c2, false, false);
        }

        // LIF layer 2 + spike accumulation
        #pragma unroll
        for (int i = 0; i < 8; ++i) {
            float mo  = mem2[i];
            float rst = (mo > THRESH) ? THRESH : 0.0f;
            float mn  = fmaf(BETA, mo, c2[i]) - rst;
            mem2[i] = mn;
            acc[i] += (mn > THRESH) ? 1.0f : 0.0f;
        }
    }

    // ---- epilogue: acc C-layout -> out[B,8] ----
    if (col < NCLS) {
        #pragma unroll
        for (int i = 0; i < 8; ++i) {
            const int row = rowBase + i + half * 8;
            out[(size_t)row * NCLS + col] = acc[i];
        }
    }
}

extern "C" void kernel_launch(void* const* d_in, const int* in_sizes, int n_in,
                              void* d_out, int out_size, void* d_ws, size_t ws_size,
                              hipStream_t stream) {
    const float* x  = (const float*)d_in[0];
    const float* W1 = (const float*)d_in[1];
    const float* b1 = (const float*)d_in[2];
    const float* W2 = (const float*)d_in[3];
    const float* b2 = (const float*)d_in[4];
    float* out = (float*)d_out;

    const int nBlocks = BATCH / (16 * WAVES_PER_BLOCK);  // 16384/64 = 256
    snn_fused_kernel<<<nBlocks, 32 * WAVES_PER_BLOCK, 0, stream>>>(x, W1, b1, W2, b2, out);
}
